// SparseLinearRelu_65609920413834
// MI455X (gfx1250) — matrix operand order repaired and checked by reading, not compiled, not run on hardware
//
#include <hip/hip_runtime.h>

// ---------------------------------------------------------------------------
// y = relu(x @ W^T + b) via bf16x3 split-GEMM on CDNA5 WMMA (gfx1250, wave32)
//   x : [M, C] f32, W : [N, C] f32, b : [N] f32, y : [M, N] f32
//   Fast path: pre-split f32 -> bf16(hi)+bf16(lo) once into workspace, then a
//   GEMM whose hot loop is async-global->LDS staging + ds_load_b128 + WMMA.
// ---------------------------------------------------------------------------

typedef __attribute__((ext_vector_type(16))) __bf16 v16bf;
typedef __attribute__((ext_vector_type(8)))  __bf16 v8bf;
typedef __attribute__((ext_vector_type(4)))  __bf16 v4bf;
typedef __attribute__((ext_vector_type(8)))  float  v8f;
typedef __attribute__((ext_vector_type(4)))  float  v4f;
typedef __attribute__((ext_vector_type(4)))  int    v4i;

#define TM   128   // output tile rows per workgroup
#define TN   128   // output tile cols per workgroup
#define TK   32    // K-step (WMMA bf16 K)
#define LDSW 40    // padded LDS row stride (bf16): 80B = 20 banks -> conflict-free b128
#define TILE_ELEMS (128 * LDSW)   // one staged 128x32 tile (padded), bf16 elems

#define AS1 __attribute__((address_space(1)))
#define AS3 __attribute__((address_space(3)))

static __device__ __forceinline__ __bf16 bf16_hi_split(float f, float& rem) {
    unsigned u = __builtin_bit_cast(unsigned, f);
    float hi_f = __builtin_bit_cast(float, u & 0xFFFF0000u);   // exact truncation
    rem = f - hi_f;                                            // exact residual
    return __builtin_bit_cast(__bf16, (unsigned short)(u >> 16));
}
static __device__ __forceinline__ __bf16 bf16_trunc(float f) {
    unsigned u = __builtin_bit_cast(unsigned, f);
    return __builtin_bit_cast(__bf16, (unsigned short)(u >> 16));
}
static __device__ __forceinline__ v16bf lds_frag16(const __bf16* p0, const __bf16* p1) {
    v8bf a = *(const v8bf*)p0;   // ds_load_b128
    v8bf b = *(const v8bf*)p1;   // ds_load_b128
    return __builtin_shufflevector(a, b, 0,1,2,3,4,5,6,7,8,9,10,11,12,13,14,15);
}

// ---- async global->LDS (CDNA5 ASYNCcnt path), with safe fallback ----------
#if defined(__has_builtin) && __has_builtin(__builtin_amdgcn_global_load_async_to_lds_b128)
#define HAVE_ASYNC_LDS 1
#else
#define HAVE_ASYNC_LDS 0
#endif

static __device__ __forceinline__ void copy16_g2l(const __bf16* g, __bf16* l) {
#if HAVE_ASYNC_LDS
    // signature (from hipcc diagnostic): (v4i addrspace(1)*, v4i addrspace(3)*, imm offset, imm cpol)
    __builtin_amdgcn_global_load_async_to_lds_b128(
        (AS1 v4i*)g, (AS3 v4i*)l, 0, 0);
#else
    *(v8bf*)l = *(const v8bf*)g;   // global_load_b128 + ds_store_b128
#endif
}

#if defined(__has_builtin) && __has_builtin(__builtin_amdgcn_s_wait_asynccnt)
#define WAIT_ASYNC(n) __builtin_amdgcn_s_wait_asynccnt(n)
#else
#define WAIT_ASYNC(n) asm volatile("s_wait_asynccnt %0" :: "i"(n) : "memory")
#endif

// Stage one 128x32 bf16 tile: 512 x 16B chunks, 2 per thread (8 async/wave total
// over the 4 tiles of one buffer).  g points at (rowBase, kc).
static __device__ __forceinline__ void stage_tile(const __bf16* __restrict__ g,
                                                  size_t gstride, __bf16* l, int tid) {
    #pragma unroll
    for (int p = 0; p < 2; ++p) {
        const int c   = tid + 256 * p;
        const int row = c >> 2;
        const int col = (c & 3) * 8;
        copy16_g2l(g + (size_t)row * gstride + col, l + row * LDSW + col);
    }
}

// ---------------------------------------------------------------------------
// Pre-split: f32 -> bf16 hi + bf16 lo (2-term split, lo truncated so hi+lo ~ f), vector x4
// ---------------------------------------------------------------------------
__global__ __launch_bounds__(256)
void split_f32_bf16x2(const float* __restrict__ src, __bf16* __restrict__ hi,
                      __bf16* __restrict__ lo)
{
    const size_t e = ((size_t)blockIdx.x * 256 + threadIdx.x) * 4;
    v4f v = *(const v4f*)(src + e);
    v4bf h, l;
    #pragma unroll
    for (int j = 0; j < 4; ++j) {
        float r;
        h[j] = bf16_hi_split(v[j], r);
        l[j] = bf16_trunc(r);
    }
    *(v4bf*)(hi + e) = h;
    *(v4bf*)(lo + e) = l;
}

// ---------------------------------------------------------------------------
// Main GEMM on pre-split bf16 inputs: async-staged, double-buffered LDS
// ---------------------------------------------------------------------------
__global__ __launch_bounds__(256)
void gemm_bf16x3_async(const __bf16* __restrict__ XH, const __bf16* __restrict__ XL,
                       const __bf16* __restrict__ WH, const __bf16* __restrict__ WL,
                       const float* __restrict__ Bv, float* __restrict__ Y,
                       int M, int C, int N)
{
    // 2 buffers x 4 tiles (XH, XL, WH, WL) x (128 x LDSW) bf16 = 80 KB
    __shared__ __align__(16) __bf16 smem[2 * 4 * TILE_ELEMS];

    const int tid  = threadIdx.x;
    const int lane = tid & 31;
    const int wave = tid >> 5;

    // N-fastest grid: consecutive blocks sweep N for one M row so the
    // 67 MB wh+wl slice stays L2-resident across the sweep.
    const int bN = blockIdx.x * TN;
    const int bM = blockIdx.y * TM;

    // 2x4 wave grid; each wave owns 64 rows x 32 cols (4x2 WMMA tiles)
    const int wM = (wave >> 2) * 64;
    const int wN = (wave & 3) * 32;

    const int half16 = (lane < 16) ? 0 : 1;
    const int lrow16 = lane & 15;

    v8f acc[4][2] = {};

    const __bf16* gXH = XH + (size_t)bM * C;
    const __bf16* gXL = XL + (size_t)bM * C;
    const __bf16* gWH = WH + (size_t)bN * C;
    const __bf16* gWL = WL + (size_t)bN * C;

    const int nk = C / TK;

    // prologue: stage buffer 0
    {
        __bf16* b0 = smem;
        stage_tile(gXH, C, b0 + 0 * TILE_ELEMS, tid);
        stage_tile(gXL, C, b0 + 1 * TILE_ELEMS, tid);
        stage_tile(gWH, C, b0 + 2 * TILE_ELEMS, tid);
        stage_tile(gWL, C, b0 + 3 * TILE_ELEMS, tid);
    }

    for (int i = 0; i < nk; ++i) {
        const int cur = i & 1;
        __bf16* bufC = smem + cur * 4 * TILE_ELEMS;

        if (i + 1 < nk) {
            // stage next K-step into the other buffer, then drain current's 8
            __bf16* bufNx = smem + (1 - cur) * 4 * TILE_ELEMS;
            const int kc = (i + 1) * TK;
            stage_tile(gXH + kc, C, bufNx + 0 * TILE_ELEMS, tid);
            stage_tile(gXL + kc, C, bufNx + 1 * TILE_ELEMS, tid);
            stage_tile(gWH + kc, C, bufNx + 2 * TILE_ELEMS, tid);
            stage_tile(gWL + kc, C, bufNx + 3 * TILE_ELEMS, tid);
            WAIT_ASYNC(8);
        } else {
            WAIT_ASYNC(0);
        }
        __syncthreads();   // whole tile visible to all waves

        const __bf16* sXh = bufC + 0 * TILE_ELEMS;
        const __bf16* sXl = bufC + 1 * TILE_ELEMS;
        const __bf16* sWh = bufC + 2 * TILE_ELEMS;
        const __bf16* sWl = bufC + 3 * TILE_ELEMS;

        // A (16x32 bf16): lanes 0-15 -> M=lane, K {0..7,16..23}; lanes 16-31 -> K {8..15,24..31}
        v16bf ah[4], al[4];
        #pragma unroll
        for (int m = 0; m < 4; ++m) {
            const int r  = wM + m * 16 + lrow16;
            const int kb = half16 * 8;
            ah[m] = lds_frag16(&sXh[r * LDSW + kb], &sXh[r * LDSW + kb + 16]);
            al[m] = lds_frag16(&sXl[r * LDSW + kb], &sXl[r * LDSW + kb + 16]);
        }
        // B (32x16 bf16): lanes 0-15 -> N=lane, K {0..7,16..23}; lanes 16-31 -> K {8..15,24..31}
        v16bf bh[2], bl[2];
        #pragma unroll
        for (int n = 0; n < 2; ++n) {
            const int r  = wN + n * 16 + lrow16;
            const int kb = half16 * 8;
            bh[n] = lds_frag16(&sWh[r * LDSW + kb], &sWh[r * LDSW + kb + 16]);
            bl[n] = lds_frag16(&sWl[r * LDSW + kb], &sWl[r * LDSW + kb + 16]);
        }

        // bf16x3 split products: hi*hi + hi*lo + lo*hi, f32 accumulate
        #pragma unroll
        for (int m = 0; m < 4; ++m) {
            #pragma unroll
            for (int n = 0; n < 2; ++n) {
                acc[m][n] = __builtin_amdgcn_wmma_f32_16x16x32_bf16(
                    false, ah[m], false, bh[n], (short)0, acc[m][n], false, false);
                acc[m][n] = __builtin_amdgcn_wmma_f32_16x16x32_bf16(
                    false, ah[m], false, bl[n], (short)0, acc[m][n], false, false);
                acc[m][n] = __builtin_amdgcn_wmma_f32_16x16x32_bf16(
                    false, al[m], false, bh[n], (short)0, acc[m][n], false, false);
            }
        }
        __syncthreads();   // all waves done reading bufC before it is re-staged
    }

    // epilogue: bias + relu + store.  C/D layout: VGPR v -> M = v + 8*half16, N = lane&15
    #pragma unroll
    for (int n = 0; n < 2; ++n) {
        const int col = bN + wN + n * 16 + lrow16;
        const float bias_v = Bv[col];
        #pragma unroll
        for (int m = 0; m < 4; ++m) {
            const int row0 = bM + wM + m * 16 + half16 * 8;
            #pragma unroll
            for (int v = 0; v < 8; ++v) {
                float r = acc[m][n][v] + bias_v;
                r = fmaxf(r, 0.0f);
                Y[(size_t)(row0 + v) * N + col] = r;
            }
        }
    }
}

// ---------------------------------------------------------------------------
// Fallback: fused in-kernel split (round-1 kernel) if workspace is too small
// ---------------------------------------------------------------------------
__global__ __launch_bounds__(256)
void sparse_linear_relu_fused(const float* __restrict__ X,
                              const float* __restrict__ W,
                              const float* __restrict__ Bv,
                              float* __restrict__ Y,
                              int M, int C, int N)
{
    __shared__ __bf16 sXh[TM * LDSW];
    __shared__ __bf16 sXl[TM * LDSW];
    __shared__ __bf16 sWh[TN * LDSW];
    __shared__ __bf16 sWl[TN * LDSW];

    const int tid  = threadIdx.x;
    const int lane = tid & 31;
    const int wave = tid >> 5;

    const int bN = blockIdx.x * TN;
    const int bM = blockIdx.y * TM;

    const int wM = (wave >> 2) * 64;
    const int wN = (wave & 3) * 32;

    const int lcol = (tid & 7) * 4;
    const int lrow = tid >> 3;

    const int half16 = (lane < 16) ? 0 : 1;
    const int lrow16 = lane & 15;

    v4f xr[4], wr[4];
    v8f acc[4][2] = {};

    const float* xp = X + (size_t)(bM + lrow) * C + lcol;
    const float* wp = W + (size_t)(bN + lrow) * C + lcol;

    #pragma unroll
    for (int i = 0; i < 4; ++i) {
        xr[i] = *(const v4f*)(xp + (size_t)(32 * i) * C);
        wr[i] = *(const v4f*)(wp + (size_t)(32 * i) * C);
    }

    for (int kc = 0; kc < C; kc += TK) {
        #pragma unroll
        for (int i = 0; i < 4; ++i) {
            const int o = (lrow + 32 * i) * LDSW + lcol;
            v4bf hv, lv;
            #pragma unroll
            for (int j = 0; j < 4; ++j) {
                float r;
                hv[j] = bf16_hi_split(xr[i][j], r);
                lv[j] = bf16_trunc(r);
            }
            *(v4bf*)&sXh[o] = hv;
            *(v4bf*)&sXl[o] = lv;
            #pragma unroll
            for (int j = 0; j < 4; ++j) {
                float r;
                hv[j] = bf16_hi_split(wr[i][j], r);
                lv[j] = bf16_trunc(r);
            }
            *(v4bf*)&sWh[o] = hv;
            *(v4bf*)&sWl[o] = lv;
        }
        __syncthreads();

        if (kc + TK < C) {
            #pragma unroll
            for (int i = 0; i < 4; ++i) {
                xr[i] = *(const v4f*)(xp + (size_t)(32 * i) * C + (kc + TK));
                wr[i] = *(const v4f*)(wp + (size_t)(32 * i) * C + (kc + TK));
            }
        }

        v16bf ah[4], al[4];
        #pragma unroll
        for (int m = 0; m < 4; ++m) {
            const int r  = wM + m * 16 + lrow16;
            const int kb = half16 * 8;
            ah[m] = lds_frag16(&sXh[r * LDSW + kb], &sXh[r * LDSW + kb + 16]);
            al[m] = lds_frag16(&sXl[r * LDSW + kb], &sXl[r * LDSW + kb + 16]);
        }
        v16bf bh[2], bl[2];
        #pragma unroll
        for (int n = 0; n < 2; ++n) {
            const int r  = wN + n * 16 + lrow16;
            const int kb = half16 * 8;
            bh[n] = lds_frag16(&sWh[r * LDSW + kb], &sWh[r * LDSW + kb + 16]);
            bl[n] = lds_frag16(&sWl[r * LDSW + kb], &sWl[r * LDSW + kb + 16]);
        }

        #pragma unroll
        for (int m = 0; m < 4; ++m) {
            #pragma unroll
            for (int n = 0; n < 2; ++n) {
                acc[m][n] = __builtin_amdgcn_wmma_f32_16x16x32_bf16(
                    false, ah[m], false, bh[n], (short)0, acc[m][n], false, false);
                acc[m][n] = __builtin_amdgcn_wmma_f32_16x16x32_bf16(
                    false, ah[m], false, bl[n], (short)0, acc[m][n], false, false);
                acc[m][n] = __builtin_amdgcn_wmma_f32_16x16x32_bf16(
                    false, al[m], false, bh[n], (short)0, acc[m][n], false, false);
            }
        }
        __syncthreads();
    }

    #pragma unroll
    for (int n = 0; n < 2; ++n) {
        const int col = bN + wN + n * 16 + lrow16;
        const float bias_v = Bv[col];
        #pragma unroll
        for (int m = 0; m < 4; ++m) {
            const int row0 = bM + wM + m * 16 + half16 * 8;
            #pragma unroll
            for (int v = 0; v < 8; ++v) {
                float r = acc[m][n][v] + bias_v;
                r = fmaxf(r, 0.0f);
                Y[(size_t)(row0 + v) * N + col] = r;
            }
        }
    }
}

// ---------------------------------------------------------------------------
extern "C" void kernel_launch(void* const* d_in, const int* in_sizes, int n_in,
                              void* d_out, int out_size, void* d_ws, size_t ws_size,
                              hipStream_t stream) {
    const float* x = (const float*)d_in[0];   // [M, C]
    const float* w = (const float*)d_in[1];   // [N, C]
    const float* b = (const float*)d_in[2];   // [N]
    float* y = (float*)d_out;                 // [M, N]

    const int N = in_sizes[2];                // 4096
    const int C = in_sizes[1] / N;            // 4096
    const int M = in_sizes[0] / C;            // 8192

    const size_t MC = (size_t)M * C;
    const size_t NC = (size_t)N * C;
    const size_t need = 2 * (MC + NC) * sizeof(__bf16);   // xh,xl,wh,wl

    dim3 grid(N / TN, M / TM);                // N-fastest for L2 reuse of W

    if (ws_size >= need) {
        __bf16* xh = (__bf16*)d_ws;
        __bf16* xl = xh + MC;
        __bf16* wh = xl + MC;
        __bf16* wl = wh + NC;
        split_f32_bf16x2<<<dim3((unsigned)(MC / 1024)), dim3(256), 0, stream>>>(x, xh, xl);
        split_f32_bf16x2<<<dim3((unsigned)(NC / 1024)), dim3(256), 0, stream>>>(w, wh, wl);
        gemm_bf16x3_async<<<grid, dim3(256), 0, stream>>>(xh, xl, wh, wl, b, y, M, C, N);
    } else {
        sparse_linear_relu_fused<<<grid, dim3(256), 0, stream>>>(x, w, b, y, M, C, N);
    }
}
